// set_values_88656714925367
// MI455X (gfx1250) — compile-verified, matrix operand order checked by reading, not run on hardware
//
#include <hip/hip_runtime.h>
#include <hip/hip_bf16.h>

// ---------------- problem constants (from reference) ----------------
#define TT   8
#define NB   4
#define CIN  64
#define HID  64
#define CIN2 128                      // concat(input_t, h)
#define HH   128
#define WW   128

#define KS_SEQ   (CIN * 9)            // 576  (k' = tap*64 + cin)
#define KS_CELL  (CIN2 * 9)           // 1152 (k' = tap*128 + cin2)

#define TILE_W   64                   // pixels per workgroup (one row segment)
#define COLS     (TILE_W + 2)         // 66 halo columns
#define ROWS     3
#define CPAD_CELL 136                 // 128 + 8 pad: lane stride 272B -> bank-conflict-free b128
#define CPAD_SEQ  72                  // 64 + 8 pad:  lane stride 144B

#define TILES_PER_ROW (WW / TILE_W)   // 2
#define NTILES (NB * HH * TILES_PER_ROW) // 1024
#define HSIZE  ((size_t)NB * HID * HH * WW)     // 4,194,304 floats

typedef __attribute__((ext_vector_type(16))) __bf16 v16bf;
typedef __attribute__((ext_vector_type(8)))  float  v8f;

union FragB {
    uint4 u[2];
    v16bf v;
};

__device__ __forceinline__ unsigned short bf16_bits(float f) {
    __bf16 b = (__bf16)f;
    return __builtin_bit_cast(unsigned short, b);
}

__device__ __forceinline__ float sigmoidf_(float x) {
    return 1.0f / (1.0f + __expf(-x));
}

// Load a 16x32 bf16 WMMA fragment (A or B 16-bit lane layout, ISA 05_wmma 7.12.2):
// lane half h=0 holds K {0..7,16..23}, h=1 holds K {8..15,24..31}; the two
// 8-element K runs are contiguous -> two 16B loads at +0 and +16 bf16.
__device__ __forceinline__ FragB load_frag_lds(const unsigned short* base) {
    FragB f;
    f.u[0] = *reinterpret_cast<const uint4*>(base);
    f.u[1] = *reinterpret_cast<const uint4*>(base + 16);
    return f;
}
__device__ __forceinline__ FragB load_frag_glb(const __bf16* base) {
    FragB f;
    f.u[0] = *reinterpret_cast<const uint4*>(base);
    f.u[1] = *reinterpret_cast<const uint4*>(base + 16);
    return f;
}

// ---------------- prep kernels ----------------
// Re-layout weights fp32 [M][cin][3][3] -> bf16 [M][tap*C + cin] (tap-major K).
__global__ void prep_weights_kernel(const float* __restrict__ wseq,
                                    unsigned short* __restrict__ wseq_bf, int n1,
                                    const float* __restrict__ wcell,
                                    unsigned short* __restrict__ wcell_bf, int n2) {
    int i = blockIdx.x * blockDim.x + threadIdx.x;
    if (i < n1) {
        int m = i / KS_SEQ, r = i - m * KS_SEQ;
        int cin = r / 9, tap = r - cin * 9;
        wseq_bf[m * KS_SEQ + tap * CIN + cin] = bf16_bits(wseq[i]);
    }
    if (i < n2) {
        int m = i / KS_CELL, r = i - m * KS_CELL;
        int cin = r / 9, tap = r - cin * 9;
        wcell_bf[m * KS_CELL + tap * CIN2 + cin] = bf16_bits(wcell[i]);
    }
}

__global__ void zero2_kernel(float* __restrict__ a, float* __restrict__ b, int n) {
    int i = blockIdx.x * blockDim.x + threadIdx.x;
    if (i < n) { a[i] = 0.0f; b[i] = 0.0f; }
}

// ---------------- conv_seq: 3x3, 64 -> 64, 9-tap shifted GEMM ----------------
// 8 wave32; wave = (mtile = wv&3, pixel-half = wv>>2); 64-pixel row tile.
__global__ __launch_bounds__(256)
void conv_seq_kernel(const float* __restrict__ x,            // [B,64,H,W] slice
                     const unsigned short* __restrict__ wbf, // [64][576] bf16, tap-major
                     const float* __restrict__ bias,
                     float* __restrict__ out)                // [B,64,H,W]
{
    __shared__ __align__(16) unsigned short raw[ROWS * COLS * CPAD_SEQ]; // 27.8 KB
    const int tid  = threadIdx.x;
    const int lane = tid & 31;
    const int wv   = tid >> 5;
    const int mt   = wv & 3;
    const int ph   = wv >> 2;

    const int p   = blockIdx.x;
    const int tpi = HH * TILES_PER_ROW;
    const int b   = p / tpi;
    const int rem = p - b * tpi;
    const int y   = rem / TILES_PER_ROW;
    const int x0  = (rem - y * TILES_PER_ROW) * TILE_W;

    // fill raw halo tile: [ry 0..2][c 0..65][cin], channel-major with pad
    for (int e = tid; e < CIN * ROWS * COLS; e += 256) {
        int cin = e / (ROWS * COLS);
        int r   = e - cin * (ROWS * COLS);
        int ry  = r / COLS;
        int c   = r - ry * COLS;
        int yy = y + ry - 1, xx = x0 + c - 1;
        float v = 0.0f;
        if ((unsigned)yy < (unsigned)HH && (unsigned)xx < (unsigned)WW)
            v = x[(((size_t)b * CIN + cin) * HH + yy) * WW + xx];
        raw[(ry * COLS + c) * CPAD_SEQ + cin] = bf16_bits(v);
    }
    __syncthreads();

    const __bf16* W = reinterpret_cast<const __bf16*>(wbf);
    const int n  = lane & 15;
    const int hh = lane >> 4;
    const __bf16* wrow = W + (size_t)(mt * 16 + n) * KS_SEQ + 8 * hh;

    v8f acc0 = {}, acc1 = {};
    #pragma unroll
    for (int tap = 0; tap < 9; ++tap) {
        const int dy = tap / 3, dx = tap - (tap / 3) * 3;
        #pragma unroll
        for (int cb = 0; cb < CIN / 32; ++cb) {
            const int slot = (dy * COLS + dx + ph * 32 + n);
            const unsigned short* lb = raw + slot * CPAD_SEQ + cb * 32 + 8 * hh;
            FragB fb0 = load_frag_lds(lb);
            FragB fb1 = load_frag_lds(lb + 16 * CPAD_SEQ);
            FragB fa  = load_frag_glb(wrow + tap * CIN + cb * 32);
            acc0 = __builtin_amdgcn_wmma_f32_16x16x32_bf16(false, fa.v, false, fb0.v, (short)0, acc0, false, false);
            acc1 = __builtin_amdgcn_wmma_f32_16x16x32_bf16(false, fa.v, false, fb1.v, (short)0, acc1, false, false);
        }
    }

    #pragma unroll
    for (int s = 0; s < 2; ++s) {
        const v8f a = s ? acc1 : acc0;
        const int xx = x0 + ph * 32 + s * 16 + n;
        #pragma unroll
        for (int r = 0; r < 8; ++r) {
            int ch = mt * 16 + r + 8 * hh;
            out[(((size_t)b * HID + ch) * HH + y) * WW + xx] = a[r] + bias[ch];
        }
    }
}

// ---------------- fused cell: 3x3 conv (128->256 gates) + LSTM update ----------------
// 8 wave32; wave = (gate-slice = wv&3, pixel-half = wv>>2).
__global__ __launch_bounds__(256)
void cell_step_kernel(const float* __restrict__ in_t,          // [B,64,H,W]
                      const float* __restrict__ h_prev,        // [B,64,H,W]
                      const unsigned short* __restrict__ wbf,  // [256][1152] bf16, tap-major
                      const float* __restrict__ bias,          // [256]
                      float* __restrict__ cbuf,                // in/out
                      float* __restrict__ h_next,
                      float* __restrict__ h_final,
                      int write_final)
{
    __shared__ __align__(16) unsigned short raw[ROWS * COLS * CPAD_CELL]; // 52.6 KB
    const int tid  = threadIdx.x;
    const int lane = tid & 31;
    const int wv   = tid >> 5;
    const int wg   = wv & 3;     // 16-channel slice of each gate
    const int ph   = wv >> 2;    // which 32-pixel half

    const int p   = blockIdx.x;
    const int tpi = HH * TILES_PER_ROW;
    const int b   = p / tpi;
    const int rem = p - b * tpi;
    const int y   = rem / TILES_PER_ROW;
    const int x0  = (rem - y * TILES_PER_ROW) * TILE_W;

    // fill raw halo tile over concat(input_t, h_prev)
    for (int e = tid; e < CIN2 * ROWS * COLS; e += 256) {
        int cin2 = e / (ROWS * COLS);
        int r    = e - cin2 * (ROWS * COLS);
        int ry   = r / COLS;
        int c    = r - ry * COLS;
        int yy = y + ry - 1, xx = x0 + c - 1;
        float v = 0.0f;
        if ((unsigned)yy < (unsigned)HH && (unsigned)xx < (unsigned)WW) {
            const float* src = (cin2 < HID) ? in_t : h_prev;
            int cc = (cin2 < HID) ? cin2 : (cin2 - HID);
            v = src[(((size_t)b * HID + cc) * HH + yy) * WW + xx];
        }
        raw[(ry * COLS + c) * CPAD_CELL + cin2] = bf16_bits(v);
    }
    __syncthreads();

    const __bf16* W = reinterpret_cast<const __bf16*>(wbf);
    const int n  = lane & 15;
    const int hh = lane >> 4;
    const int mloc = wg * 16 + n;
    const __bf16* wr0 = W + (size_t)(0 * HID + mloc) * KS_CELL + 8 * hh;
    const __bf16* wr1 = W + (size_t)(1 * HID + mloc) * KS_CELL + 8 * hh;
    const __bf16* wr2 = W + (size_t)(2 * HID + mloc) * KS_CELL + 8 * hh;
    const __bf16* wr3 = W + (size_t)(3 * HID + mloc) * KS_CELL + 8 * hh;

    v8f aI0 = {}, aI1 = {}, aF0 = {}, aF1 = {};
    v8f aG0 = {}, aG1 = {}, aO0 = {}, aO1 = {};

    for (int tap = 0; tap < 9; ++tap) {
        const int dy = tap / 3, dx = tap - (tap / 3) * 3;
        #pragma unroll
        for (int cb = 0; cb < CIN2 / 32; ++cb) {
            const int slot = (dy * COLS + dx + ph * 32 + n);
            const unsigned short* lb = raw + slot * CPAD_CELL + cb * 32 + 8 * hh;
            FragB fb0 = load_frag_lds(lb);
            FragB fb1 = load_frag_lds(lb + 16 * CPAD_CELL);
            const int kcol = tap * CIN2 + cb * 32;

            FragB fa = load_frag_glb(wr0 + kcol);
            aI0 = __builtin_amdgcn_wmma_f32_16x16x32_bf16(false, fa.v, false, fb0.v, (short)0, aI0, false, false);
            aI1 = __builtin_amdgcn_wmma_f32_16x16x32_bf16(false, fa.v, false, fb1.v, (short)0, aI1, false, false);
            fa = load_frag_glb(wr1 + kcol);
            aF0 = __builtin_amdgcn_wmma_f32_16x16x32_bf16(false, fa.v, false, fb0.v, (short)0, aF0, false, false);
            aF1 = __builtin_amdgcn_wmma_f32_16x16x32_bf16(false, fa.v, false, fb1.v, (short)0, aF1, false, false);
            fa = load_frag_glb(wr2 + kcol);
            aG0 = __builtin_amdgcn_wmma_f32_16x16x32_bf16(false, fa.v, false, fb0.v, (short)0, aG0, false, false);
            aG1 = __builtin_amdgcn_wmma_f32_16x16x32_bf16(false, fa.v, false, fb1.v, (short)0, aG1, false, false);
            fa = load_frag_glb(wr3 + kcol);
            aO0 = __builtin_amdgcn_wmma_f32_16x16x32_bf16(false, fa.v, false, fb0.v, (short)0, aO0, false, false);
            aO1 = __builtin_amdgcn_wmma_f32_16x16x32_bf16(false, fa.v, false, fb1.v, (short)0, aO1, false, false);
        }
    }

    #pragma unroll
    for (int s = 0; s < 2; ++s) {
        const v8f vI = s ? aI1 : aI0;
        const v8f vF = s ? aF1 : aF0;
        const v8f vG = s ? aG1 : aG0;
        const v8f vO = s ? aO1 : aO0;
        const int xx = x0 + ph * 32 + s * 16 + n;
        #pragma unroll
        for (int r = 0; r < 8; ++r) {
            int ch = wg * 16 + r + 8 * hh;   // 0..63
            size_t idx = (((size_t)b * HID + ch) * HH + y) * WW + xx;
            float iv = vI[r] + bias[0 * HID + ch];
            float fv = vF[r] + bias[1 * HID + ch];
            float gv = vG[r] + bias[2 * HID + ch];
            float ov = vO[r] + bias[3 * HID + ch];
            float cp = cbuf[idx];
            float cn = sigmoidf_(fv) * cp + sigmoidf_(iv) * tanhf(gv);
            float hn = sigmoidf_(ov) * tanhf(cn);
            cbuf[idx]   = cn;
            h_next[idx] = hn;
            if (write_final) h_final[idx] = hn;
        }
    }
}

// ---------------- launch ----------------
extern "C" void kernel_launch(void* const* d_in, const int* in_sizes, int n_in,
                              void* d_out, int out_size, void* d_ws, size_t ws_size,
                              hipStream_t stream) {
    const float* xinp   = (const float*)d_in[0];
    const float* W_seq  = (const float*)d_in[1];
    const float* b_seq  = (const float*)d_in[2];
    const float* W_cell = (const float*)d_in[3];
    const float* b_cell = (const float*)d_in[4];

    float* out     = (float*)d_out;
    float* h_final = out;                 // [B,HID,H,W]
    float* xout    = out + HSIZE;         // [T,B,HID,H,W]

    char* ws = (char*)d_ws;
    const int NWSEQ  = HID * KS_SEQ;          // 36864
    const int NWCELL = 4 * HID * KS_CELL;     // 294912
    unsigned short* wseq_bf  = (unsigned short*)ws;
    unsigned short* wcell_bf = (unsigned short*)(ws + (size_t)NWSEQ * 2);
    size_t off = ((size_t)NWSEQ + NWCELL) * 2;
    off = (off + 255) & ~(size_t)255;
    float* h0   = (float*)(ws + off); off += HSIZE * sizeof(float);
    float* h1   = (float*)(ws + off); off += HSIZE * sizeof(float);
    float* cbuf = (float*)(ws + off); off += HSIZE * sizeof(float);

    prep_weights_kernel<<<(NWCELL + 255) / 256, 256, 0, stream>>>(
        W_seq, wseq_bf, NWSEQ, W_cell, wcell_bf, NWCELL);
    zero2_kernel<<<((int)HSIZE + 255) / 256, 256, 0, stream>>>(h0, cbuf, (int)HSIZE);

    float* hp[2] = { h0, h1 };
    for (int t = 0; t < TT; ++t) {
        const float* x_t = xinp + (size_t)t * NB * CIN * HH * WW;
        float* in_t = xout + (size_t)t * NB * HID * HH * WW;
        conv_seq_kernel<<<NTILES, 256, 0, stream>>>(x_t, wseq_bf, b_seq, in_t);
        cell_step_kernel<<<NTILES, 256, 0, stream>>>(
            in_t, hp[t & 1], wcell_bf, b_cell, cbuf, hp[(t + 1) & 1],
            h_final, (t == TT - 1) ? 1 : 0);
    }
}
